// OriLinearGNN_6846177869846
// MI455X (gfx1250) — matrix operand order repaired and checked by reading, not compiled, not run on hardware
//
#include <hip/hip_runtime.h>
#include <hip/hip_bf16.h>

// ---------------------------------------------------------------------------
// OriLinearGNN on MI455X (gfx1250, wave32, WMMA)
//
// Phase 1 (WMMA): per-edge A = tanh([emb[src]||emb[dst]] @ Wxi + bxi) * (mu/S)/dg
//                 stored f16 ([E,64], 102.4 MB -> L2-resident for the steps),
//                 and b = tanh(emb[src] @ Wrou + brou) stored f32 ([E,8]).
//                 Weights staged in LDS TRANSPOSED (f16, conflict-free padded
//                 strides) -> every WMMA B fragment is two ds_load_b128s.
//                 tanh epilogue uses hardware v_tanh_f32 when available
//                 (TRANS op, co-executes with the WMMA pipe).
// Phase 2: T=4 steps, thread per (edge,i): dot(A[e,i,:], states[src]) + b,
//          scatter via global_atomic_add_f32 (segment_sum by X_Node).
// Phase 3: logits = [W_emb || states] @ Wout + bout, softmax over 3.
//
// Workspace: E*64 halfs + E*8 floats + 2*N*8 floats ~= 131.2 MB.
// ---------------------------------------------------------------------------

typedef __attribute__((ext_vector_type(16))) _Float16 v16h;
typedef __attribute__((ext_vector_type(8)))  _Float16 v8h;
typedef __attribute__((ext_vector_type(8)))  float    v8f;

#define GNN_LN 64
#define GNN_S  8

#if __has_builtin(__builtin_amdgcn_tanhf)
__device__ __forceinline__ float fast_tanh(float x) {
  return __builtin_amdgcn_tanhf(x);      // v_tanh_f32 (gfx1250 TRANS op)
}
#else
__device__ __forceinline__ float fast_tanh(float x) { return tanhf(x); }
#endif

// LDS column strides (halfs). Chosen so per-lane b128 reads are 16B-aligned
// and conflict-free on the 64-bank LDS:
//   136 halfs = 272 B -> lane n starts at DWORD n*68 ; n*68 mod 64 = n*4
//   72  halfs = 144 B -> lane n starts at DWORD n*36 ; all 16 distinct
#define WXI_STRIDE  136
#define WROU_STRIDE 72

// ---------------------------------------------------------------------------
// Phase 1: edge precompute with WMMA. Block = 256 threads = 8 waves,
// each wave owns 16 consecutive edges; block covers 128 edges.
// ---------------------------------------------------------------------------
__global__ __launch_bounds__(256) void edge_precompute_wmma(
    const int* __restrict__ xnode, const int* __restrict__ xneis,
    const float* __restrict__ dg,
    const float* __restrict__ Wemb,         // [N,64]
    const float* __restrict__ Wxi,          // [128,64] row-major (k, n)
    const float* __restrict__ bxi,          // [64]
    const float* __restrict__ Wrou,         // [64,8]  row-major (k, n)
    const float* __restrict__ brou,         // [8]
    _Float16* __restrict__ A_ws,            // [E,64] f16
    float* __restrict__ b_ws,               // [E,8]  f32
    int E)
{
  // Transposed f16 weight staging: sWxiT[n*WXI_STRIDE + k] = Wxi[k,n]
  __shared__ _Float16 sWxiT[64 * WXI_STRIDE];    // 17.0 KB
  __shared__ _Float16 sWrouT[16 * WROU_STRIDE];  // 2.25 KB (cols 8..15 zero)
  __shared__ float    sbxi[64];
  __shared__ float    sbrou[8];

  const int tid = threadIdx.x;
  for (int i = tid; i < 128 * 64; i += 256) {
    const int k = i >> 6, n = i & 63;             // read Wxi contiguously
    sWxiT[n * WXI_STRIDE + k] = (_Float16)Wxi[i];
  }
  for (int i = tid; i < 16 * 64; i += 256) {
    const int n = i >> 6, k = i & 63;
    sWrouT[n * WROU_STRIDE + k] =
        (n < 8) ? (_Float16)Wrou[k * 8 + n] : (_Float16)0.0f;
  }
  if (tid < 64) sbxi[tid] = bxi[tid];
  if (tid < 8)  sbrou[tid] = brou[tid];
  __syncthreads();

  const int wave = tid >> 5;
  const int lane = tid & 31;
  const int half = lane >> 4;     // 0: lanes 0-15, 1: lanes 16-31
  const int mrow = lane & 15;
  const int edgeBase = blockIdx.x * 128 + wave * 16;

  // Clamp loads so EXEC stays all-1s across the WMMAs; stores predicated later.
  int myEdge = edgeBase + mrow;
  if (myEdge >= E) myEdge = E - 1;
  const float* nodeRow = Wemb + (size_t)xnode[myEdge] * GNN_LN;
  const float* neisRow = Wemb + (size_t)xneis[myEdge] * GNN_LN;

  // A-matrix fragments, f16 16x32 layout:
  //   lanes 0-15 : M=lane,    K local = {0..7} then {8..15}
  //   lanes 16-31: M=lane-16, K local = {16..23} then {24..31}
  // K-chunks 0,1 come from emb[src] (k=0..63), chunks 2,3 from emb[dst].
  // All segment starts are multiples of 8 floats (32 B) off a 256 B-aligned
  // row, so float4 loads are aligned -> global_load_b128.
  v16h afrag[4];
  #pragma unroll
  for (int kc = 0; kc < 4; ++kc) {
    const float* src = (kc < 2) ? nodeRow : neisRow;
    const int kb = (kc & 1) * 32;
    const int s0 = kb + (half ? 16 : 0);
    const int s1 = kb + 8 + (half ? 16 : 0);
    const float4 a0 = ((const float4*)(src + s0))[0];
    const float4 a1 = ((const float4*)(src + s0))[1];
    const float4 a2 = ((const float4*)(src + s1))[0];
    const float4 a3 = ((const float4*)(src + s1))[1];
    v16h f;
    f[0]  = (_Float16)a0.x; f[1]  = (_Float16)a0.y;
    f[2]  = (_Float16)a0.z; f[3]  = (_Float16)a0.w;
    f[4]  = (_Float16)a1.x; f[5]  = (_Float16)a1.y;
    f[6]  = (_Float16)a1.z; f[7]  = (_Float16)a1.w;
    f[8]  = (_Float16)a2.x; f[9]  = (_Float16)a2.y;
    f[10] = (_Float16)a2.z; f[11] = (_Float16)a2.w;
    f[12] = (_Float16)a3.x; f[13] = (_Float16)a3.y;
    f[14] = (_Float16)a3.z; f[15] = (_Float16)a3.w;
    afrag[kc] = f;
  }

  // Main GEMM: [16 x 128] @ [128 x 64] -> 4 accumulators of 16x16 f32.
  // B fragment, f16 32x16: lanes 0-15 hold N=lane (K local 0..15),
  // lanes 16-31 hold N=lane-16 (K local 16..31) -> contiguous in sWxiT.
  v8f acc[4] = {};
  #pragma unroll
  for (int kc = 0; kc < 4; ++kc) {
    const int krow0 = kc * 32 + (half ? 16 : 0);
    #pragma unroll
    for (int nt = 0; nt < 4; ++nt) {
      const int n = nt * 16 + mrow;
      const _Float16* colp = &sWxiT[n * WXI_STRIDE + krow0];
      const v8h lo = *(const v8h*)colp;        // ds_load_b128, conflict-free
      const v8h hi = *(const v8h*)(colp + 8);  // ds_load_b128, conflict-free
      v16h bf;
      #pragma unroll
      for (int j = 0; j < 8; ++j) { bf[j] = lo[j]; bf[8 + j] = hi[j]; }
      acc[nt] = __builtin_amdgcn_wmma_f32_16x16x32_f16(
          false, afrag[kc], false, bf, (short)0, acc[nt], false, false);
    }
  }

  // b GEMM: [16 x 64] @ [64 x 16(pad)] using k-chunks 0,1 of the same A frags.
  v8f accB = {};
  #pragma unroll
  for (int kc = 0; kc < 2; ++kc) {
    const int krow0 = kc * 32 + (half ? 16 : 0);
    const _Float16* colp = &sWrouT[mrow * WROU_STRIDE + krow0];
    const v8h lo = *(const v8h*)colp;
    const v8h hi = *(const v8h*)(colp + 8);
    v16h bf;
    #pragma unroll
    for (int j = 0; j < 8; ++j) { bf[j] = lo[j]; bf[8 + j] = hi[j]; }
    accB = __builtin_amdgcn_wmma_f32_16x16x32_f16(
        false, afrag[kc], false, bf, (short)0, accB, false, false);
  }

  // Epilogue: D layout (f32 16x16): lane -> N = (lane&15) [+tile],
  // vgpr r -> M = r + 8*half.
  const float scaleBase = 0.9f / 8.0f;   // MU / S
  #pragma unroll
  for (int r = 0; r < 8; ++r) {
    const int m = r + half * 8;
    const int e = edgeBase + m;
    if (e < E) {
      const float invd = scaleBase / dg[e];
      #pragma unroll
      for (int nt = 0; nt < 4; ++nt) {
        const int n = nt * 16 + mrow;
        const float v = fast_tanh(acc[nt][r] + sbxi[n]);
        A_ws[(size_t)e * 64 + n] = (_Float16)(v * invd);
      }
      if (mrow < 8) {
        b_ws[(size_t)e * 8 + mrow] = fast_tanh(accB[r] + sbrou[mrow]);
      }
    }
  }
}

// ---------------------------------------------------------------------------
// Zero fill
// ---------------------------------------------------------------------------
__global__ __launch_bounds__(256) void zero_f32(float* __restrict__ p, int n) {
  int t = blockIdx.x * blockDim.x + threadIdx.x;
  if (t < n) p[t] = 0.0f;
}

// ---------------------------------------------------------------------------
// Phase 2: one propagation step. Thread per (edge, i):
//   out[i] = sum_j A[e,i,j] * states_in[src[e], j] + b[e,i]
//   atomicAdd into states_out[src[e], i]   (segment_sum by X_Node)
// A_ws is L2-resident (102.4 MB < 192 MB L2).
// ---------------------------------------------------------------------------
__global__ __launch_bounds__(256) void gnn_step(
    const int* __restrict__ xnode,
    const _Float16* __restrict__ A_ws,
    const float* __restrict__ b_ws,
    const float* __restrict__ s_in,
    float* __restrict__ s_out,
    int E)
{
  int t = blockIdx.x * blockDim.x + threadIdx.x;
  if (t >= E * GNN_S) return;
  const int e = t >> 3;
  const int i = t & 7;
  const int node = xnode[e];
  // 32 B state row, 32 B-aligned -> two global_load_b128.
  const float4* hp = (const float4*)(s_in + (size_t)node * GNN_S);
  const float4 h0 = hp[0];
  const float4 h1 = hp[1];
  // 16 B of f16 A, 16 B-aligned -> one global_load_b128.
  const v8h a = *(const v8h*)(A_ws + (size_t)e * 64 + i * 8);
  float acc = b_ws[(size_t)e * GNN_S + i];
  acc += (float)a[0] * h0.x + (float)a[1] * h0.y +
         (float)a[2] * h0.z + (float)a[3] * h0.w +
         (float)a[4] * h1.x + (float)a[5] * h1.y +
         (float)a[6] * h1.z + (float)a[7] * h1.w;
  atomicAdd(s_out + (size_t)node * GNN_S + i, acc);
}

// ---------------------------------------------------------------------------
// Phase 3: logits = [W_emb || states] @ Wout + bout, softmax over 3 classes.
// ---------------------------------------------------------------------------
__global__ __launch_bounds__(128) void out_softmax(
    const float* __restrict__ Wemb,
    const float* __restrict__ states,
    const float* __restrict__ Wout,     // [72,3]
    const float* __restrict__ bout,     // [3]
    float* __restrict__ out,            // [N,3]
    int Nn)
{
  int n = blockIdx.x * blockDim.x + threadIdx.x;
  if (n >= Nn) return;
  float l0 = bout[0], l1 = bout[1], l2 = bout[2];
  const float* er = Wemb + (size_t)n * GNN_LN;
  #pragma unroll 8
  for (int k = 0; k < GNN_LN; ++k) {
    const float x = er[k];
    l0 += x * Wout[k * 3 + 0];
    l1 += x * Wout[k * 3 + 1];
    l2 += x * Wout[k * 3 + 2];
  }
  const float* sr = states + (size_t)n * GNN_S;
  #pragma unroll
  for (int k = 0; k < GNN_S; ++k) {
    const float x = sr[k];
    const int kk = GNN_LN + k;
    l0 += x * Wout[kk * 3 + 0];
    l1 += x * Wout[kk * 3 + 1];
    l2 += x * Wout[kk * 3 + 2];
  }
  const float m = fmaxf(l0, fmaxf(l1, l2));
  const float e0 = __expf(l0 - m), e1 = __expf(l1 - m), e2 = __expf(l2 - m);
  const float inv = 1.0f / (e0 + e1 + e2);
  out[(size_t)n * 3 + 0] = e0 * inv;
  out[(size_t)n * 3 + 1] = e1 * inv;
  out[(size_t)n * 3 + 2] = e2 * inv;
}

// ---------------------------------------------------------------------------
// kernel_launch
// inputs: 0 X_Node(i32,E) 1 X_Neis(i32,E) 2 dg_list(f32,E) 3 W_emb(f32,N*64)
//         4 Wxi(f32,128*64) 5 bxi(f32,64) 6 Wrou(f32,64*8) 7 brou(f32,8)
//         8 Wout(f32,72*3) 9 bout(f32,3)
// out: softmax(f32, N*3)
// ---------------------------------------------------------------------------
extern "C" void kernel_launch(void* const* d_in, const int* in_sizes, int n_in,
                              void* d_out, int out_size, void* d_ws, size_t ws_size,
                              hipStream_t stream) {
  const int*   xnode = (const int*)d_in[0];
  const int*   xneis = (const int*)d_in[1];
  const float* dg    = (const float*)d_in[2];
  const float* Wemb  = (const float*)d_in[3];
  const float* Wxi   = (const float*)d_in[4];
  const float* bxi   = (const float*)d_in[5];
  const float* Wrou  = (const float*)d_in[6];
  const float* brou  = (const float*)d_in[7];
  const float* Wout  = (const float*)d_in[8];
  const float* bout  = (const float*)d_in[9];

  const int E  = in_sizes[0];
  const int Nn = in_sizes[3] / GNN_LN;

  // Workspace carve-up (all offsets 4B-aligned).
  _Float16* A_ws = (_Float16*)d_ws;                       // E*64 halfs
  float*    b_ws = (float*)(A_ws + (size_t)E * 64);       // E*8 floats
  float*    st0  = b_ws + (size_t)E * GNN_S;              // N*8 floats
  float*    st1  = st0 + (size_t)Nn * GNN_S;              // N*8 floats

  // Phase 1: WMMA edge precompute (16 edges/wave, 128 edges/block).
  edge_precompute_wmma<<<(E + 127) / 128, 256, 0, stream>>>(
      xnode, xneis, dg, Wemb, Wxi, bxi, Wrou, brou, A_ws, b_ws, E);

  // states = 0
  const int nsElems = Nn * GNN_S;
  zero_f32<<<(nsElems + 255) / 256, 256, 0, stream>>>(st0, nsElems);

  // Phase 2: T=4 propagation steps (ping-pong state buffers).
  float* sin  = st0;
  float* sout = st1;
  const int stepThreads = E * GNN_S;
  for (int t = 0; t < 4; ++t) {
    zero_f32<<<(nsElems + 255) / 256, 256, 0, stream>>>(sout, nsElems);
    gnn_step<<<(stepThreads + 255) / 256, 256, 0, stream>>>(
        xnode, A_ws, b_ws, sin, sout, E);
    float* tmp = sin; sin = sout; sout = tmp;
  }

  // Phase 3: output logits + softmax.
  out_softmax<<<(Nn + 127) / 128, 128, 0, stream>>>(
      Wemb, sin, Wout, bout, (float*)d_out, Nn);
}
